// CausalSelfAttention_26499948216800
// MI455X (gfx1250) — matrix operand order repaired and checked by reading, not compiled, not run on hardware
//
#include <hip/hip_runtime.h>

// ---------------------------------------------------------------------------
// Causal self-attention for MI455X (gfx1250, wave32, WMMA).
// Compute-bound (~103 GFLOP vs ~50MB HBM traffic) -> all matmuls through
// v_wmma_f32_16x16x32_f16 with f32 accumulation. Flash-style attention with
// K/V tiles staged in LDS and shared by 8 waves (8x L2 traffic reduction),
// double-buffered through registers. Needs 48MB+16KB of d_ws.
// ---------------------------------------------------------------------------

#define BB 2
#define LL 2048
#define CC 1024
#define HH 16
#define DD 64
#define MM (BB * LL)   // 4096 rows (b,l)

typedef __attribute__((ext_vector_type(16))) _Float16 v16h;
typedef __attribute__((ext_vector_type(8)))  float    v8f;
typedef __attribute__((ext_vector_type(4)))  unsigned v4u;  // trivial, union-safe
typedef __attribute__((ext_vector_type(4)))  float    v4f;

union F16x16 { v16h v; _Float16 h[16]; unsigned u[8]; v4u q[2]; };
union F32x8v { v4f v4[2]; float f[8]; };

template <bool V> struct BoolC { static constexpr bool value = V; };

__device__ inline v8f wmma_f16(v16h a, v16h b, v8f c) {
  // D(16x16 f32) = A(16x32 f16) * B(32x16 f16) + C
  return __builtin_amdgcn_wmma_f32_16x16x32_f16(false, a, false, b,
                                                (short)0, c, false, false);
}

__device__ inline v8f vzero8() {
  v8f v = {0.f, 0.f, 0.f, 0.f, 0.f, 0.f, 0.f, 0.f};
  return v;
}

// ---------------------------------------------------------------------------
// fp32 -> fp16 elementwise conversion
// ---------------------------------------------------------------------------
__global__ void cvt_f32_f16(const float* __restrict__ s,
                            _Float16* __restrict__ d, int n) {
  int i = (blockIdx.x * blockDim.x + threadIdx.x) * 4;
  if (i + 3 < n) {
    float4 v = *(const float4*)(s + i);
    d[i + 0] = (_Float16)v.x;
    d[i + 1] = (_Float16)v.y;
    d[i + 2] = (_Float16)v.z;
    d[i + 3] = (_Float16)v.w;
  } else {
    for (int j = i; j < n; ++j) d[j] = (_Float16)s[j];
  }
}

// int64 key-mask -> additive f32 bias (0 or -1e30)
__global__ void mask_to_bias(const long long* __restrict__ m,
                             float* __restrict__ bias, int n) {
  int i = blockIdx.x * blockDim.x + threadIdx.x;
  if (i < n) bias[i] = (m[i] != 0) ? 0.f : -1e30f;
}

// ---------------------------------------------------------------------------
// Tiled WMMA GEMM:  Cmat[M x N] = A[M x K(row-major)] * W[N x K]^T
// M=4096, N=K=1024. Block = 256 threads = 8 waves in 4(M) x 2(N) layout;
// each wave computes a 32x32 tile as 2x2 WMMA fragments; block tile 128x64.
// Double-buffered via registers; LDS stride 40 halves = 80B (16B aligned).
// MODE: 0 = Q f16 per-head [b,h,l,d] scaled by 1/sqrt(D)
//       1 = K f16 per-head [b,h,l,d]
//       2 = V f16 per-head TRANSPOSED [b,h,d,l]
//       3 = f32 row-major [m,n]  (final output projection)
// ---------------------------------------------------------------------------
template <int MODE>
__global__ __launch_bounds__(256)
void gemm_xwT(const _Float16* __restrict__ A, const _Float16* __restrict__ W,
              _Float16* __restrict__ outh, float* __restrict__ outf) {
  __shared__ _Float16 As[128 * 40];
  __shared__ _Float16 Bs[64 * 40];

  const int t    = threadIdx.x;
  const int w    = t >> 5;
  const int lane = t & 31;
  const int lq   = lane & 15;   // N col in B/C frags, M row in A frags
  const int hi   = lane >> 4;   // half-wave select
  const int wm   = w & 3;       // wave M quadrant (0..3)
  const int wn   = w >> 2;      // wave N half (0..1)
  const int m0   = blockIdx.y * 128;
  const int n0   = blockIdx.x * 64;

  auto loadA = [&](int k0, v4u* ra) {
    for (int i = 0; i < 2; ++i) {
      int idx = i * 256 + t, row = idx >> 2, part = idx & 3;
      ra[i] = *(const v4u*)(A + (size_t)(m0 + row) * CC + k0 + part * 8);
    }
  };
  auto loadB = [&](int k0, v4u* rb) {
    int row = t >> 2, part = t & 3;
    rb[0] = *(const v4u*)(W + (size_t)(n0 + row) * CC + k0 + part * 8);
  };
  auto storeLDS = [&](const v4u* ra, const v4u* rb) {
    for (int i = 0; i < 2; ++i) {
      int idx = i * 256 + t, row = idx >> 2, part = idx & 3;
      *(v4u*)(As + row * 40 + part * 8) = ra[i];
    }
    int row = t >> 2, part = t & 3;
    *(v4u*)(Bs + row * 40 + part * 8) = rb[0];
  };

  v8f acc[2][2];
  acc[0][0] = vzero8(); acc[0][1] = vzero8();
  acc[1][0] = vzero8(); acc[1][1] = vzero8();

  v4u ra[2], rb[1];
  loadA(0, ra); loadB(0, rb);
  storeLDS(ra, rb);

  for (int k0 = 0; k0 < CC; k0 += 32) {
    __syncthreads();                      // LDS tile ready
    const bool hasNext = (k0 + 32) < CC;  // uniform across block
    v4u na[2], nb[1];
    if (hasNext) { loadA(k0 + 32, na); loadB(k0 + 32, nb); }

    // A-frag (16x32 f16): lane row = lq; halves [8hi..8hi+7],[16+8hi..+7]
    F16x16 af[2], bf[2];
    for (int fi = 0; fi < 2; ++fi) {
      const _Float16* rp = As + (wm * 32 + fi * 16 + lq) * 40;
      af[fi].q[0] = *(const v4u*)(rp + 8 * hi);
      af[fi].q[1] = *(const v4u*)(rp + 16 + 8 * hi);
    }
    // B-frag (32x16 f16): lane col = lq; halves [8hi..8hi+7],[16+8hi..+7]
    for (int fj = 0; fj < 2; ++fj) {
      const _Float16* rp = Bs + (wn * 32 + fj * 16 + lq) * 40;
      bf[fj].q[0] = *(const v4u*)(rp + 8 * hi);
      bf[fj].q[1] = *(const v4u*)(rp + 16 + 8 * hi);
    }
    for (int fi = 0; fi < 2; ++fi)
      for (int fj = 0; fj < 2; ++fj)
        acc[fi][fj] = wmma_f16(af[fi].v, bf[fj].v, acc[fi][fj]);

    __syncthreads();                      // everyone done reading LDS
    if (hasNext) storeLDS(na, nb);
  }

  // Store. C/D frag: VGPR r, lanes0-15: M=r, N=lane; lanes16-31: M=r+8.
  for (int fi = 0; fi < 2; ++fi) {
    for (int fj = 0; fj < 2; ++fj) {
      for (int r = 0; r < 8; ++r) {
        int m = m0 + wm * 32 + fi * 16 + r + 8 * hi;
        int n = n0 + wn * 32 + fj * 16 + lq;
        float val = acc[fi][fj][r];
        int b = m / LL, li = m % LL;
        int h = n / DD, d = n % DD;
        if (MODE == 0) {
          outh[(((size_t)(b * HH + h)) * LL + li) * DD + d] =
              (_Float16)(val * 0.125f);   // fold 1/sqrt(64)
        } else if (MODE == 1) {
          outh[(((size_t)(b * HH + h)) * LL + li) * DD + d] = (_Float16)val;
        } else if (MODE == 2) {
          outh[(((size_t)(b * HH + h)) * DD + d) * LL + li] = (_Float16)val;
        } else {
          outf[(size_t)m * CC + n] = val;
        }
      }
    }
  }
}

// ---------------------------------------------------------------------------
// Flash attention. Block = 256 threads = 8 waves, each wave owns 16 query
// rows of one (b,h); the block's 128 queries SHARE each 32-key K/V tile
// through LDS (double-buffered, register-staged -> global loads overlap
// WMMAs). Per wave:
//   S^T = K_tile(16xD) @ Q^T(Dx16): each lane owns one query column, so the
//   online softmax reduces in-lane + one shfl_xor(16), and P^T drops into
//   the B-fragment of O^T += V^T @ P^T with two xor-16 exchanges per row.
// Exactly one tile per wave straddles the causal diagonal (wave-uniform
// branch); waves past their range skip compute but keep the barriers.
// ---------------------------------------------------------------------------
__global__ __launch_bounds__(256)
void attn_flash(const _Float16* __restrict__ Qh, const _Float16* __restrict__ Kh,
                const _Float16* __restrict__ Vt,
                const float* __restrict__ biasAll,
                _Float16* __restrict__ Oh) {
  // LDS tiles: K 32x64 (stride 72 halves = 144B, 16B aligned), V^T 64x32
  // (stride 40 halves = 80B), both double-buffered. ~19KB total.
  __shared__ _Float16 Ks[2][32 * 72];
  __shared__ _Float16 Vs[2][64 * 40];

  const int t    = threadIdx.x;
  const int w    = t >> 5;
  const int lane = t & 31;
  const int lq   = lane & 15;
  const int hi   = lane >> 4;
  const int bh   = blockIdx.y;           // b*H + h
  const int b    = bh >> 4;
  const int h    = bh & 15;
  const int blockq = blockIdx.x * 128;
  const int qbase  = blockq + w * 16;
  const int qg     = qbase + lq;         // this lane's query row

  const size_t headOff = (size_t)bh * LL * DD;   // same size for [l][d] & [d][l]
  const float* bias = biasAll + (size_t)b * LL;

  // Staging slots: K tile 32 rows x 8 v4u -> row=t>>3, part=t&7
  //                V tile 64 rows x 4 v4u -> row=t>>2, part=t&3
  const int krow = t >> 3, kpart = t & 7;
  const int vrow = t >> 2, vpart = t & 3;
  auto loadKV = [&](int kbase, v4u& nk, v4u& nv) {
    nk = *(const v4u*)(Kh + headOff + (size_t)(kbase + krow) * DD + kpart * 8);
    nv = *(const v4u*)(Vt + headOff + (size_t)vrow * LL + kbase + vpart * 8);
  };
  auto storeKV = [&](int buf, v4u nk, v4u nv) {
    *(v4u*)(&Ks[buf][krow * 72 + kpart * 8]) = nk;
    *(v4u*)(&Vs[buf][vrow * 40 + vpart * 8]) = nv;
  };

  // Q as two B-fragments (N=q, K=d): halves j*32 + [8hi..8hi+7],[16+8hi..+7]
  F16x16 qf[2];
  {
    const _Float16* qrow = Qh + headOff + (size_t)qg * DD;
    for (int j = 0; j < 2; ++j) {
      qf[j].q[0] = *(const v4u*)(qrow + j * 32 + 8 * hi);
      qf[j].q[1] = *(const v4u*)(qrow + j * 32 + 16 + 8 * hi);
    }
  }

  v8f o[4];
  for (int i = 0; i < 4; ++i) o[i] = vzero8();
  float mrun = -1e30f, lrun = 0.f;

  const int nfull = qbase >> 5;          // this wave's tiles below the diagonal
  const int ntb   = blockIdx.x * 4 + 4;  // block-uniform tile count

  auto tile = [&](int buf, int kbase, auto tag) {
    constexpr bool CAUSAL = decltype(tag)::value;

    // Token-mask bias, contiguous per lane: keys kbase+8*hi+r / +16
    F32x8v bw0, bw1;
    {
      const float* bp = bias + kbase + 8 * hi;
      bw0.v4[0] = *(const v4f*)(bp);
      bw0.v4[1] = *(const v4f*)(bp + 4);
      bw1.v4[0] = *(const v4f*)(bp + 16);
      bw1.v4[1] = *(const v4f*)(bp + 20);
    }

    // S^T for two 16-key subtiles, K-dim D=64 in two chunks of 32
    v8f t0 = vzero8(), t1 = vzero8();
    const _Float16* Kp0 = &Ks[buf][lq * 72];
    const _Float16* Kp1 = &Ks[buf][(16 + lq) * 72];
    for (int dc = 0; dc < 2; ++dc) {
      F16x16 kf0, kf1;
      kf0.q[0] = *(const v4u*)(Kp0 + dc * 32 + 8 * hi);
      kf0.q[1] = *(const v4u*)(Kp0 + dc * 32 + 16 + 8 * hi);
      kf1.q[0] = *(const v4u*)(Kp1 + dc * 32 + 8 * hi);
      kf1.q[1] = *(const v4u*)(Kp1 + dc * 32 + 16 + 8 * hi);
      t0 = wmma_f16(kf0.v, qf[dc].v, t0);
      t1 = wmma_f16(kf1.v, qf[dc].v, t1);
    }

    // Mask + online softmax. Lane holds keys kbase+r+8*hi (t0) and +16 (t1).
    float e0[8], e1[8];
    float mx = -1e30f;
    for (int r = 0; r < 8; ++r) {
      float s0 = t0[r] + bw0.f[r];
      float s1 = t1[r] + bw1.f[r];
      if (CAUSAL) {
        int k0i = kbase + r + 8 * hi;
        s0 = (k0i <= qg)      ? s0 : -1e30f;
        s1 = (k0i + 16 <= qg) ? s1 : -1e30f;
      }
      e0[r] = s0; e1[r] = s1;
      mx = fmaxf(mx, fmaxf(s0, s1));
    }
    mx = fmaxf(mx, __shfl_xor(mx, 16, 32));
    float mnew = fmaxf(mrun, mx);
    float sc   = __expf(mrun - mnew);
    float psum = 0.f;
    for (int r = 0; r < 8; ++r) {
      e0[r] = __expf(e0[r] - mnew);
      e1[r] = __expf(e1[r] - mnew);
      psum += e0[r] + e1[r];
    }
    psum += __shfl_xor(psum, 16, 32);
    lrun = lrun * sc + psum;
    mrun = mnew;

    // P^T B-fragment (32 keys x 16 q): two xor-16 exchanges per row.
    F16x16 pf;
    for (int r = 0; r < 8; ++r) {
      float sw0 = __shfl_xor(e0[r], 16, 32);
      float sw1 = __shfl_xor(e1[r], 16, 32);
      pf.h[r]     = (_Float16)(hi ? sw1 : e0[r]);
      pf.h[r + 8] = (_Float16)(hi ? e1[r] : sw0);
    }

    // Rescale accumulators, then O^T += V^T(16d x 32k) @ P^T(32k x 16q)
    for (int i = 0; i < 4; ++i)
      for (int r = 0; r < 8; ++r) o[i][r] *= sc;
    for (int i = 0; i < 4; ++i) {
      const _Float16* Vp = &Vs[buf][(i * 16 + lq) * 40];
      F16x16 vf;
      vf.q[0] = *(const v4u*)(Vp + 16 * hi);
      vf.q[1] = *(const v4u*)(Vp + 16 * hi + 8);
      o[i] = wmma_f16(vf.v, pf.v, o[i]);
    }
  };

  // Prologue: stage tile 0
  {
    v4u nk, nv;
    loadKV(0, nk, nv);
    storeKV(0, nk, nv);
  }

  int buf = 0;
  for (int tIdx = 0; tIdx < ntb; ++tIdx) {
    __syncthreads();                       // LDS tile `buf` ready
    const bool hasNext = (tIdx + 1) < ntb; // block-uniform
    v4u nk, nv;
    if (hasNext) loadKV((tIdx + 1) * 32, nk, nv);

    if (tIdx < nfull)       tile(buf, tIdx * 32, BoolC<false>{});  // wave-uniform
    else if (tIdx == nfull) tile(buf, tIdx * 32, BoolC<true>{});
    // tIdx > nfull: this wave is done; keep the barriers.

    __syncthreads();                       // everyone done reading `buf`
    if (hasNext) storeKV(buf ^ 1, nk, nv);
    buf ^= 1;
  }

  // Normalize and emit O as f16 row-major [b,l,c] for the output projection.
  float inv = 1.f / ((lrun > 0.f) ? lrun : 1.f);
  for (int i = 0; i < 4; ++i) {
    for (int r = 0; r < 8; ++r) {
      int d = i * 16 + r + 8 * hi;
      Oh[((size_t)(b * LL) + qg) * CC + h * DD + d] = (_Float16)(o[i][r] * inv);
    }
  }
}

// ---------------------------------------------------------------------------
// Launch
// ---------------------------------------------------------------------------
extern "C" void kernel_launch(void* const* d_in, const int* in_sizes, int n_in,
                              void* d_out, int out_size, void* d_ws, size_t ws_size,
                              hipStream_t stream) {
  const float*     x     = (const float*)d_in[0];
  const long long* amask = (const long long*)d_in[1];
  const float*     Wq    = (const float*)d_in[2];
  const float*     Wk    = (const float*)d_in[3];
  const float*     Wv    = (const float*)d_in[4];
  const float*     Wp    = (const float*)d_in[5];
  float*           out   = (float*)d_out;

  // Workspace layout: 24M f16 (48MB) + 4096 f32 bias (16KB).
  _Float16* ws  = (_Float16*)d_ws;
  const size_t SZ_X = (size_t)MM * CC;       // 4M
  const size_t SZ_W = (size_t)CC * CC;       // 1M
  _Float16* xh  = ws;
  _Float16* Wqh = xh  + SZ_X;
  _Float16* Wkh = Wqh + SZ_W;
  _Float16* Wvh = Wkh + SZ_W;
  _Float16* Wph = Wvh + SZ_W;
  _Float16* Qh  = Wph + SZ_W;                // [b,h,l,d]
  _Float16* Kh  = Qh  + SZ_X;                // [b,h,l,d]
  _Float16* Vt  = Kh  + SZ_X;                // [b,h,d,l]
  _Float16* Oh  = Vt  + SZ_X;                // [b,l,c]
  float*    bias = (float*)(Oh + SZ_X);      // [b,l]

  // 1) f32 -> f16 conversions + mask bias
  {
    int n = (int)SZ_X;
    cvt_f32_f16<<<(n / 4 + 255) / 256, 256, 0, stream>>>(x, xh, n);
    n = (int)SZ_W;
    int g = (n / 4 + 255) / 256;
    cvt_f32_f16<<<g, 256, 0, stream>>>(Wq, Wqh, n);
    cvt_f32_f16<<<g, 256, 0, stream>>>(Wk, Wkh, n);
    cvt_f32_f16<<<g, 256, 0, stream>>>(Wv, Wvh, n);
    cvt_f32_f16<<<g, 256, 0, stream>>>(Wp, Wph, n);
    mask_to_bias<<<(BB * LL + 255) / 256, 256, 0, stream>>>(amask, bias, BB * LL);
  }

  // 2) Projections (WMMA GEMM): Q scaled, K row-major, V transposed
  dim3 gg(CC / 64, MM / 128);   // (16, 32)
  gemm_xwT<0><<<gg, 256, 0, stream>>>(xh, Wqh, Qh, nullptr);
  gemm_xwT<1><<<gg, 256, 0, stream>>>(xh, Wkh, Kh, nullptr);
  gemm_xwT<2><<<gg, 256, 0, stream>>>(xh, Wvh, Vt, nullptr);

  // 3) Flash attention: grid (L/128, B*H), K/V tiles shared via LDS
  attn_flash<<<dim3(LL / 128, BB * HH), 256, 0, stream>>>(Qh, Kh, Vt, bias, Oh);

  // 4) Output projection, f32 out
  gemm_xwT<3><<<gg, 256, 0, stream>>>(Oh, Wph, nullptr, out);
}
